// VanillaRNN_16509854286536
// MI455X (gfx1250) — matrix operand order repaired and checked by reading, not compiled, not run on hardware
//
#include <hip/hip_runtime.h>

typedef __attribute__((ext_vector_type(16))) __bf16 v16bf;
typedef __attribute__((ext_vector_type(8)))  __bf16 v8bf;
typedef __attribute__((ext_vector_type(8)))  float  v8f;

#define B_  256
#define T_  512
#define H_  1024
#define C_  10
#define NWG 32          // persistent workgroups; each owns 32 output columns
#define KF  (H_/32)     // 32 K-fragments of depth 32

__device__ __forceinline__ unsigned short f2bf(float f) {
    unsigned u = __builtin_bit_cast(unsigned, f);
    u += 0x7FFFu + ((u >> 16) & 1u);          // round-to-nearest-even
    return (unsigned short)(u >> 16);
}
__device__ __forceinline__ float bf2f(unsigned short s) {
    unsigned u = ((unsigned)s) << 16;
    return __builtin_bit_cast(float, u);
}

// ---- init: zero h0 (bf16) and the grid-barrier counter --------------------
__global__ void rnn_init(unsigned* __restrict__ syncw, unsigned short* __restrict__ h0) {
    int idx = blockIdx.x * blockDim.x + threadIdx.x;
    if (idx < B_ * H_) h0[idx] = 0;           // bf16 zero
    if (idx == 0) { syncw[0] = 0; syncw[1] = 0; }
}

// ---- persistent recurrence kernel -----------------------------------------
__global__ __launch_bounds__(256)
void rnn_steps(const float* __restrict__ x,      // [B,T]
               const float* __restrict__ whx,    // [H,1]
               const float* __restrict__ whh,    // [H,H]
               const float* __restrict__ bias_h, // [H]
               unsigned short* __restrict__ hbuf,// 2 x [B,H] bf16 (double buffer)
               unsigned* __restrict__ syncw)
{
    // whh slice (32 cols x 1024 k, bf16) pre-swizzled into B-fragment order: 64 KB
    __shared__ __attribute__((aligned(32))) unsigned lds_b[KF][2][32][8];
    __shared__ float lds_x[B_];

    const int tid   = threadIdx.x;
    const int lane  = tid & 31;
    const int wave  = tid >> 5;
    const int half  = lane >> 4;
    const int ln    = lane & 15;
    const int jbase = blockIdx.x * 32;

    // One-time: pack whh rows [jbase, jbase+32) to bf16 in exact B-fragment
    // layout: lds_b[kf][nt][lane][v] = pack(whh[j][k], whh[j][k+1]),
    // j = jbase + nt*16 + lane%16, k = kf*32 + (lane/16)*16 + 2v.
    for (int idx = tid; idx < KF * 2 * 32 * 8; idx += blockDim.x) {
        int v  = idx & 7;
        int l  = (idx >> 3) & 31;
        int nt = (idx >> 8) & 1;
        int kf = idx >> 9;
        int j  = jbase + nt * 16 + (l & 15);
        int k  = kf * 32 + (l >> 4) * 16 + 2 * v;
        unsigned lo = f2bf(whh[j * H_ + k]);
        unsigned hi = f2bf(whh[j * H_ + k + 1]);
        lds_b[kf][nt][l][v] = lo | (hi << 16);
    }

    const int   j0  = jbase + ln;         // lane's output column, N-tile 0
    const int   j1  = jbase + 16 + ln;    // lane's output column, N-tile 1
    const float wj0 = whx[j0];
    const float bj0 = bias_h[j0];
    const float wj1 = whx[j1];
    const float bj1 = bias_h[j1];
    const int   b0  = wave * 32;          // this wave's 32 batch rows (2 M-tiles)
    const int   r0  = b0 + ln;            // A-row, M-tile 0
    const int   r1  = b0 + 16 + ln;       // A-row, M-tile 1

    __syncthreads();

    for (int t = 0; t < T_; ++t) {
        const __bf16*   hprev = (const __bf16*)(hbuf + (t & 1) * (B_ * H_));
        unsigned short* hnext = hbuf + ((t + 1) & 1) * (B_ * H_);

        lds_x[tid] = x[tid * T_ + t];       // stage x[:, t]
        __syncthreads();

        v8f acc00 = {};   // (M-tile 0, N-tile 0)
        v8f acc01 = {};   // (M-tile 0, N-tile 1)
        v8f acc10 = {};   // (M-tile 1, N-tile 0)
        v8f acc11 = {};   // (M-tile 1, N-tile 1)
        for (int kf = 0; kf < KF; ++kf) {
            const int k0 = kf * 32 + half * 16;
            // A fragments: lane layout = one contiguous 16-element K run per lane
            v8bf a0lo = *(const v8bf*)(hprev + r0 * H_ + k0);
            v8bf a0hi = *(const v8bf*)(hprev + r0 * H_ + k0 + 8);
            v8bf a1lo = *(const v8bf*)(hprev + r1 * H_ + k0);
            v8bf a1hi = *(const v8bf*)(hprev + r1 * H_ + k0 + 8);
            v16bf a0 = __builtin_shufflevector(a0lo, a0hi, 0,1,2,3,4,5,6,7,8,9,10,11,12,13,14,15);
            v16bf a1 = __builtin_shufflevector(a1lo, a1hi, 0,1,2,3,4,5,6,7,8,9,10,11,12,13,14,15);
            // B fragments: pre-swizzled in LDS, 2x ds_load_b128 each
            v8bf b0lo = *(const v8bf*)&lds_b[kf][0][lane][0];
            v8bf b0hi = *(const v8bf*)&lds_b[kf][0][lane][4];
            v8bf b1lo = *(const v8bf*)&lds_b[kf][1][lane][0];
            v8bf b1hi = *(const v8bf*)&lds_b[kf][1][lane][4];
            v16bf bm0 = __builtin_shufflevector(b0lo, b0hi, 0,1,2,3,4,5,6,7,8,9,10,11,12,13,14,15);
            v16bf bm1 = __builtin_shufflevector(b1lo, b1hi, 0,1,2,3,4,5,6,7,8,9,10,11,12,13,14,15);

            acc00 = __builtin_amdgcn_wmma_f32_16x16x32_bf16(false, a0, false, bm0, (short)0, acc00, false, false);
            acc01 = __builtin_amdgcn_wmma_f32_16x16x32_bf16(false, a0, false, bm1, (short)0, acc01, false, false);
            acc10 = __builtin_amdgcn_wmma_f32_16x16x32_bf16(false, a1, false, bm0, (short)0, acc10, false, false);
            acc11 = __builtin_amdgcn_wmma_f32_16x16x32_bf16(false, a1, false, bm1, (short)0, acc11, false, false);
        }

        // epilogue: h = tanh(acc + whx[j]*x[b,t] + bias[j]), store bf16
        #pragma unroll
        for (int r = 0; r < 8; ++r) {
            int m  = r + half * 8;
            int ba = b0 + m;          // rows of M-tile 0
            int bb = b0 + 16 + m;     // rows of M-tile 1
            float xa = lds_x[ba];
            float xb = lds_x[bb];
            hnext[ba * H_ + j0] = f2bf(tanhf(acc00[r] + wj0 * xa + bj0));
            hnext[ba * H_ + j1] = f2bf(tanhf(acc01[r] + wj1 * xa + bj1));
            hnext[bb * H_ + j0] = f2bf(tanhf(acc10[r] + wj0 * xb + bj0));
            hnext[bb * H_ + j1] = f2bf(tanhf(acc11[r] + wj1 * xb + bj1));
        }

        // grid-wide barrier: monotonic arrival counter (reset by rnn_init each launch)
        __threadfence();
        __syncthreads();
        if (tid == 0) {
            atomicAdd(syncw, 1u);
            const unsigned target = (unsigned)NWG * (unsigned)(t + 1);
            while (*(volatile unsigned*)syncw < target) {
                __builtin_amdgcn_s_sleep(1);
            }
        }
        __syncthreads();
        __threadfence();
    }
}

// ---- final projection: p = h_T @ wph^T + bias_p ---------------------------
__global__ void rnn_proj(const unsigned short* __restrict__ hlast,
                         const float* __restrict__ wph,
                         const float* __restrict__ bias_p,
                         float* __restrict__ out)
{
    int idx = blockIdx.x * blockDim.x + threadIdx.x;
    if (idx >= B_ * C_) return;
    int b = idx / C_;
    int c = idx % C_;
    float acc = bias_p[c];
    for (int k = 0; k < H_; ++k)
        acc += bf2f(hlast[b * H_ + k]) * wph[c * H_ + k];
    out[idx] = acc;
}

extern "C" void kernel_launch(void* const* d_in, const int* in_sizes, int n_in,
                              void* d_out, int out_size, void* d_ws, size_t ws_size,
                              hipStream_t stream) {
    (void)in_sizes; (void)n_in; (void)out_size; (void)ws_size;
    const float* x      = (const float*)d_in[0];
    const float* whx    = (const float*)d_in[1];
    const float* whh    = (const float*)d_in[2];
    const float* bias_h = (const float*)d_in[3];
    const float* wph    = (const float*)d_in[4];
    const float* bias_p = (const float*)d_in[5];

    unsigned*       syncw = (unsigned*)d_ws;
    unsigned short* hbuf  = (unsigned short*)((char*)d_ws + 256); // 2 x B*H bf16 = 1 MB

    rnn_init<<<(B_ * H_ + 255) / 256, 256, 0, stream>>>(syncw, hbuf);
    rnn_steps<<<NWG, 256, 0, stream>>>(x, whx, whh, bias_h, hbuf, syncw);
    // after t=511, h_last lives in buffer 0 ((511+1)&1 == 0)
    rnn_proj<<<(B_ * C_ + 255) / 256, 256, 0, stream>>>(hbuf, wph, bias_p, (float*)d_out);
}
